// Encoder_70557722738954
// MI455X (gfx1250) — matrix operand order repaired and checked by reading, not compiled, not run on hardware
//
#include <hip/hip_runtime.h>
#include <cmath>

typedef __bf16 bf16_t;
typedef __attribute__((ext_vector_type(16))) __bf16 v16bf;
typedef __attribute__((ext_vector_type(8)))  __bf16 v8bf;
typedef __attribute__((ext_vector_type(8)))  float  v8f;
typedef __attribute__((__vector_size__(4 * sizeof(int)))) int v4i;

#define SEQ   2048
#define BATCH 64
#define INDIM 300
#define KPAD  320      // K padded to multiple of 32 for 16x16x32 WMMA
#define HID   512
#define MTOT  (SEQ * BATCH)   // 131072 rows for the hoisted input GEMM

// gfx1250 async global->LDS copy (ASYNCcnt-tracked)
__device__ inline void async_copy_b128(const void* g, void* l) {
  __builtin_amdgcn_global_load_async_to_lds_b128((v4i*)g, (v4i*)l, 0, 0);
}

// ---------------------------------------------------------------------------
// pad-convert f32 [rows][cin] -> bf16 [rows][cout], zero padding k in [cin,cout)
// ---------------------------------------------------------------------------
__global__ void cvt_pad_kernel(const float* __restrict__ src, bf16_t* __restrict__ dst,
                               long long rows, int cin, int cout) {
  long long i = (long long)blockIdx.x * blockDim.x + threadIdx.x;
  if (i >= rows * (long long)cout) return;
  int k = (int)(i % cout);
  long long r = i / cout;
  float v = (k < cin) ? src[r * (long long)cin + k] : 0.0f;
  dst[i] = (bf16_t)v;
}

// ---------------------------------------------------------------------------
// Phase 1: preb[m][n] = bf16( sum_k X[m][k]*W_ih[n][k] + b_ih[n] + b_hh[n] )
// One wave computes a 64x16 output tile (4 M-tiles x 1 N-tile), K=320 (10 k-tiles).
// B tiles (W_ih) live in VGPRs for the whole wave lifetime.
// ---------------------------------------------------------------------------
__global__ void __launch_bounds__(256) gemm_pre_kernel(
    const bf16_t* __restrict__ xb,   // [MTOT][KPAD] bf16
    const bf16_t* __restrict__ wb,   // [HID][KPAD]  bf16 (row n = W_ih[n][:])
    const float* __restrict__ bih, const float* __restrict__ bhh,
    bf16_t* __restrict__ preb)       // [MTOT][HID] bf16
{
  const int lane   = threadIdx.x & 31;
  const int wave   = threadIdx.x >> 5;
  const int hiHalf = lane >> 4;       // 0: lanes 0-15, 1: lanes 16-31
  const int nlane  = lane & 15;
  const int ntile  = blockIdx.y * 8 + wave;   // 0..31
  const int n      = ntile * 16 + nlane;
  const int m0     = blockIdx.x * 64;

  // B (32x16) per-lane layout: lane holds column n, 16 consecutive K values
  // (lanes 0-15: K=kt*32..+15, lanes 16-31: K=kt*32+16..+31) -> contiguous in wb row n.
  v16bf Bt[KPAD / 32];
  const bf16_t* wrow = wb + (size_t)n * KPAD;
#pragma unroll
  for (int kt = 0; kt < KPAD / 32; ++kt)
    Bt[kt] = *(const v16bf*)(wrow + kt * 32 + hiHalf * 16);

  const float bias = bih[n] + bhh[n];

#pragma unroll
  for (int mt = 0; mt < 4; ++mt) {
    v8f c = {};
    // A (16x32) per-lane layout: row M = lane&15; lanes 0-15 hold K {0..15},
    // lanes 16-31 hold K {16..31} -> two 16B loads at +0 and +8 from base.
    const bf16_t* abase = xb + (size_t)(m0 + mt * 16 + nlane) * KPAD + hiHalf * 16;
#pragma unroll
    for (int kt = 0; kt < KPAD / 32; ++kt) {
      union { v16bf v; v8bf h[2]; } a;
      a.h[0] = *(const v8bf*)(abase + kt * 32);
      a.h[1] = *(const v8bf*)(abase + kt * 32 + 8);
      c = __builtin_amdgcn_wmma_f32_16x16x32_bf16(false, a.v, false, Bt[kt],
                                                  (short)0, c, false, false);
    }
    // C/D layout: VGPR r -> M = mt*16 + hiHalf*8 + r, N = n
    bf16_t* prow = preb + (size_t)(m0 + mt * 16 + hiHalf * 8) * HID + n;
#pragma unroll
    for (int r = 0; r < 8; ++r)
      prow[(size_t)r * HID] = (bf16_t)(c[r] + bias);
  }
}

// ---------------------------------------------------------------------------
// Phase 2: sequential scan. One persistent workgroup (16 waves) on one WGP.
// W_hh stays resident in VGPRs (each wave: 32 columns x K=512 = 32 B-tiles,
// 256 VGPRs); h lives in LDS (64KB bf16). pre[t+1] slices are staged into
// double-buffered LDS via async global->LDS copies that overlap step t's
// WMMA work (per-wave ASYNCcnt is the only sync needed: each wave stages
// exactly the slice it consumes). Per step: 2048 WMMAs, tanh, s_barrier.
// ---------------------------------------------------------------------------
__global__ void __launch_bounds__(512, 1) rnn_scan_kernel(
    const bf16_t* __restrict__ whb,  // [HID][HID] bf16 (row n = W_hh[n][:])
    const bf16_t* __restrict__ preb, // [SEQ*BATCH][HID] bf16
    float* __restrict__ out)         // [BATCH][HID] f32
{
  __shared__ __align__(32) bf16_t hbuf[BATCH * HID];        // 64 KB
  __shared__ __align__(16) bf16_t pstage[2][16][BATCH * 32]; // 2 x 64 KB staging
  const int lane   = threadIdx.x & 31;
  const int wave   = threadIdx.x >> 5;   // 0..15
  const int hiHalf = lane >> 4;
  const int nlane  = lane & 15;

  for (int i = threadIdx.x; i < BATCH * HID; i += 512)
    hbuf[i] = (bf16_t)0.0f;             // h0 = 0

  // Resident W_hh: wave owns output columns [32*wave, 32*wave+32)
  v16bf Bt[2][16];
#pragma unroll
  for (int j = 0; j < 2; ++j) {
    const int n = (wave * 2 + j) * 16 + nlane;
    const bf16_t* wrow = whb + (size_t)n * HID;
#pragma unroll
    for (int kt = 0; kt < 16; ++kt)
      Bt[j][kt] = *(const v16bf*)(wrow + kt * 32 + hiHalf * 16);
  }

  // Stage this wave's 64x32 slice of pre[tt] (4 KB) into pstage[tt&1][wave].
  auto stage_pre = [&](int tt) {
    const bf16_t* gsrc = preb + (size_t)tt * (BATCH * HID) + wave * 32;
    bf16_t* ldst = &pstage[tt & 1][wave][0];
#pragma unroll
    for (int q = 0; q < 8; ++q) {
      int s  = (q * 32 + lane) * 8;   // element index in slice (16B chunks)
      int r  = s >> 5;                // row 0..63
      int c0 = s & 31;                // col 0..31 (8-aligned)
      async_copy_b128(gsrc + (size_t)r * HID + c0, ldst + s);
    }
  };

  stage_pre(0);
  __syncthreads();

  bf16_t hnew[2][4][8];
  for (int t = 0; t < SEQ; ++t) {
    __builtin_amdgcn_s_wait_asynccnt(0);   // slice for step t is in LDS
    if (t + 1 < SEQ) stage_pre(t + 1);     // overlap staging with step t math
    if (t + 2 < SEQ)                       // deep prefetch toward L2
      __builtin_prefetch(preb + (size_t)(t + 2) * (BATCH * HID) + threadIdx.x * 64, 0, 0);

    const bf16_t* myslice = &pstage[t & 1][wave][0];
#pragma unroll
    for (int j = 0; j < 2; ++j) {
      const int n = (wave * 2 + j) * 16 + nlane;
      const int cloc = j * 16 + nlane;     // local column in slice
#pragma unroll
      for (int mt = 0; mt < 4; ++mt) {
        v8f c;
#pragma unroll
        for (int r = 0; r < 8; ++r)
          c[r] = (float)myslice[(mt * 16 + hiHalf * 8 + r) * 32 + cloc];

        const bf16_t* abase = hbuf + (mt * 16 + nlane) * HID + hiHalf * 16;
#pragma unroll
        for (int kt = 0; kt < 16; ++kt) {
          union { v16bf v; v8bf h[2]; } a;
          a.h[0] = *(const v8bf*)(abase + kt * 32);
          a.h[1] = *(const v8bf*)(abase + kt * 32 + 8);
          c = __builtin_amdgcn_wmma_f32_16x16x32_bf16(false, a.v, false, Bt[j][kt],
                                                      (short)0, c, false, false);
        }
#pragma unroll
        for (int r = 0; r < 8; ++r) hnew[j][mt][r] = (bf16_t)tanhf(c[r]);
      }
    }
    __syncthreads();   // all waves finished reading h_{t-1}
#pragma unroll
    for (int j = 0; j < 2; ++j) {
      const int n = (wave * 2 + j) * 16 + nlane;
#pragma unroll
      for (int mt = 0; mt < 4; ++mt)
#pragma unroll
        for (int r = 0; r < 8; ++r)
          hbuf[(mt * 16 + hiHalf * 8 + r) * HID + n] = hnew[j][mt][r];
    }
    __syncthreads();   // h_t visible to all waves
  }

  for (int i = threadIdx.x; i < BATCH * HID; i += 512)
    out[i] = (float)hbuf[i];
}

// ---------------------------------------------------------------------------
extern "C" void kernel_launch(void* const* d_in, const int* in_sizes, int n_in,
                              void* d_out, int out_size, void* d_ws, size_t ws_size,
                              hipStream_t stream) {
  (void)in_sizes; (void)n_in; (void)out_size; (void)ws_size;
  const float* x   = (const float*)d_in[0];   // [SEQ][BATCH][INDIM]
  const float* wih = (const float*)d_in[1];   // [HID][INDIM]
  const float* whh = (const float*)d_in[2];   // [HID][HID]
  const float* bih = (const float*)d_in[3];   // [HID]
  const float* bhh = (const float*)d_in[4];   // [HID]
  float* out = (float*)d_out;                 // [BATCH][HID]

  char* ws = (char*)d_ws;
  size_t off = 0;
  auto carve = [&](size_t bytes) -> char* {
    char* p = ws + off;
    off += (bytes + 255) & ~(size_t)255;
    return p;
  };
  bf16_t* xb   = (bf16_t*)carve((size_t)MTOT * KPAD * sizeof(bf16_t)); // 80 MiB (L2-resident)
  bf16_t* wb   = (bf16_t*)carve((size_t)HID  * KPAD * sizeof(bf16_t)); // 320 KiB
  bf16_t* whb  = (bf16_t*)carve((size_t)HID  * HID  * sizeof(bf16_t)); // 512 KiB
  bf16_t* preb = (bf16_t*)carve((size_t)MTOT * (size_t)HID * sizeof(bf16_t)); // 128 MiB

  { long long n = (long long)MTOT * KPAD;
    cvt_pad_kernel<<<(unsigned)((n + 255) / 256), 256, 0, stream>>>(x, xb, MTOT, INDIM, KPAD); }
  { long long n = (long long)HID * KPAD;
    cvt_pad_kernel<<<(unsigned)((n + 255) / 256), 256, 0, stream>>>(wih, wb, HID, INDIM, KPAD); }
  { long long n = (long long)HID * HID;
    cvt_pad_kernel<<<(unsigned)((n + 255) / 256), 256, 0, stream>>>(whh, whb, HID, HID, HID); }

  // Phase 1: hoisted input GEMM, fully parallel across the chip.
  gemm_pre_kernel<<<dim3(MTOT / 64, 4), 256, 0, stream>>>(xb, wb, bih, bhh, preb);

  // Phase 2: persistent single-WGP recurrent scan, W_hh in registers,
  // async-LDS double-buffered pre slices.
  rnn_scan_kernel<<<1, 512, 0, stream>>>(whb, preb, out);
}
